// LSTMCellQ_bmm_24163486007738
// MI455X (gfx1250) — matrix operand order repaired and checked by reading, not compiled, not run on hardware
//
#include <hip/hip_runtime.h>
#include <hip/hip_bf16.h>
#include <math.h>

// ---------------------------------------------------------------------------
// Quantized 4-block LSTM cell for gfx1250 (MI455X).
// HBM-bound on fp32 weights (~210 MB @ 23.3 TB/s ~= 9 us): each weight
// element is read exactly once, quantized to f16 in-register (the 8-bit
// quant grid k/128 is exact in f16), and consumed by v_wmma_f32_16x16x32_f16.
// A-operand tiles are DMA'd to LDS by the Tensor Data Mover
// (tensor_load_to_lds + s_wait_tensorcnt).
// ---------------------------------------------------------------------------

#define NB     4
#define BATCH  128
#define KDIM   1024
#define NCOLS  4096
#define KT     32        // WMMA K per step (f16)
#define MT     128       // M tile per workgroup (whole batch)
#define NT     64        // N tile per workgroup -> 256 workgroups

typedef __attribute__((ext_vector_type(16))) _Float16 v16h;
typedef __attribute__((ext_vector_type(8)))  _Float16 v8h;
typedef __attribute__((ext_vector_type(8)))  float    v8f;
typedef __attribute__((ext_vector_type(4)))  unsigned int v4u;
typedef __attribute__((ext_vector_type(8)))  int      v8i;
typedef __attribute__((ext_vector_type(4)))  int      v4i;

#if defined(__has_builtin)
#if __has_builtin(__builtin_amdgcn_tensor_load_to_lds) && \
    __has_builtin(__builtin_amdgcn_s_wait_tensorcnt)
#define HAVE_TDM 1
#endif
#endif
#ifndef HAVE_TDM
#define HAVE_TDM 0
#endif

// pact(x,a) == clamp(x,-a,a); then qp(.,8,a): clip to +/-127/128, snap to 1/128 grid.
__device__ __forceinline__ float qpact8(float x, float a, float inv_a) {
    float t = x * inv_a;
    t = fminf(fmaxf(t, -0.9921875f), 0.9921875f);
    return rintf(t * 128.0f) * 0.0078125f * a;
}

// weight/bias quant: qp(clip(w,-1,1), 8, 1.0)
__device__ __forceinline__ float wq8f(float w) {
    float t = fminf(fmaxf(w, -0.9921875f), 0.9921875f);
    return rintf(t * 128.0f) * 0.0078125f;
}

__device__ __forceinline__ float sigmoid_(float x) {
    return 1.0f / (1.0f + __expf(-x));
}

// ---------------------------------------------------------------------------
// Kernel 1: quantize x (replicated per block, clip a1[n]) and hx (a11[n]) to f16.
// ---------------------------------------------------------------------------
__global__ void __launch_bounds__(256)
quant_inputs_kernel(const float* __restrict__ x, const float* __restrict__ hx,
                    const float* __restrict__ a1, const float* __restrict__ a11,
                    _Float16* __restrict__ Aih, _Float16* __restrict__ Ahh) {
    int idx = blockIdx.x * blockDim.x + threadIdx.x;      // over NB*BATCH*KDIM
    int n   = idx >> 17;                                  // / (128*1024)
    int mk  = idx & ((BATCH * KDIM) - 1);
    float c1  = a1[n],  i1  = 1.0f / c1;
    float c11 = a11[n], i11 = 1.0f / c11;
    Aih[idx] = (_Float16)qpact8(x[mk],  c1,  i1);
    Ahh[idx] = (_Float16)qpact8(hx[idx], c11, i11);
}

// ---------------------------------------------------------------------------
// Kernel 2: fused dual-GEMM producing quantized gates (fp32) in workspace.
// grid = (NCOLS/NT, NB), block = 256 threads (8 waves).
// Each wave owns a 32x32 C sub-tile (2x2 WMMA accumulators, dual phase).
// ---------------------------------------------------------------------------
__global__ void __launch_bounds__(256)
gemm_gates_kernel(const _Float16* __restrict__ Aih, const _Float16* __restrict__ Ahh,
                  const float* __restrict__ Wih, const float* __restrict__ Whh,
                  const float* __restrict__ wmask,
                  const float* __restrict__ bih, const float* __restrict__ bhh,
                  const float* __restrict__ a12p, const float* __restrict__ a13p,
                  const float* __restrict__ a14p,
                  float* __restrict__ gates) {
    __shared__ _Float16 As[MT * KT];   // [m][k], 8 KB (TDM destination)
    __shared__ _Float16 Bs[NT * KT];   // [n][k] (transposed at stage), 4 KB

    const int n  = blockIdx.y;
    const int n0 = blockIdx.x * NT;

    const int tid   = threadIdx.x;
    const int lane  = tid & 31;
    const int wave  = tid >> 5;
    const int wr    = wave & 3;        // M sub-strip (32 rows)
    const int wc    = wave >> 2;       // N sub-strip (32 cols)
    const int l16   = lane & 15;
    const int lhalf = lane >> 4;       // 0 or 1 (K half selector)

    const _Float16* Aih_n = Aih + (size_t)n * BATCH * KDIM;
    const _Float16* Ahh_n = Ahh + (size_t)n * BATCH * KDIM;
    const float*    Wih_n = Wih + (size_t)n * KDIM * NCOLS;
    const float*    Whh_n = Whh + (size_t)n * KDIM * NCOLS;
    const float*    Msk_n = wmask + (size_t)n * KDIM * NCOLS;
    float*          G_n   = gates + (size_t)n * BATCH * NCOLS;

    const float a12 = a12p[n], a13 = a13p[n], a14 = a14p[n];
    const float i12 = 1.0f / a12, i13 = 1.0f / a13, i14 = 1.0f / a14;

#if HAVE_TDM
    // Low 32 bits of the generic address of an LDS object are its LDS byte
    // offset (flat->LDS aperture mapping truncates to addr[31:0]).
    const unsigned ldsA = (unsigned)(size_t)(&As[0]);
#endif

    float q1[2][2][8];                 // quantized part1 tile, kept in VGPRs

    for (int phase = 0; phase < 2; ++phase) {
        const _Float16* A_src = phase ? Ahh_n : Aih_n;
        const float*    W_src = phase ? Whh_n : Wih_n;

        v8f acc[2][2];
        #pragma unroll
        for (int mi = 0; mi < 2; ++mi)
            #pragma unroll
            for (int ni = 0; ni < 2; ++ni) { v8f z = {}; acc[mi][ni] = z; }

        for (int k0 = 0; k0 < KDIM; k0 += KT) {
#if HAVE_TDM
            // ---- A tile (128x32 f16) via Tensor Data Mover, one DMA/workgroup.
            // D# per CDNA5 ISA: 2-D tile, data_size=2B,
            // tile_dim0=KT, tile_dim1=BATCH, tensor_dim0_stride=KDIM.
            if (wave == 0) {
                unsigned long long ga =
                    (unsigned long long)(size_t)(A_src + k0);
                v4u g0;
                g0.x = 1u;                                   // count=1 (valid user D#)
                g0.y = ldsA;                                 // LDS byte address
                g0.z = (unsigned)(ga & 0xffffffffu);         // global_addr[31:0]
                g0.w = (unsigned)((ga >> 32) & 0x01ffffffu)  // global_addr[56:32]
                       | (2u << 30);                         // type=2 ("image")
                v8i g1 = {
                    (int)(1u << 16),                         // data_size=1 -> 2 bytes
                    (int)(((unsigned)KDIM & 0xffffu) << 16), // tensor_dim0[15:0]
                    (int)(((unsigned)KDIM >> 16) |
                          (((unsigned)BATCH & 0xffffu) << 16)), // dim0[31:16]|dim1[15:0]
                    (int)(((unsigned)KT) << 16),             // dim1[31:16]=0 | tile_dim0
                    (int)BATCH,                              // tile_dim1 | tile_dim2=0
                    (int)KDIM,                               // tensor_dim0_stride[31:0]
                    0,                                       // stride hi | dim1_stride lo
                    0 };
                v4i gz4 = {0, 0, 0, 0};                      // groups 2/3 unused (2-D)
                v8i gz8 = {0, 0, 0, 0, 0, 0, 0, 0};
                __builtin_amdgcn_tensor_load_to_lds(g0, g1, gz4, gz4, gz8, 0);
            }
#else
            // ---- fallback: stage A tile with vector loads ----
            #pragma unroll
            for (int r = 0; r < 2; ++r) {
                int h  = (r * 256 + tid) * 8;
                int m  = h >> 5;
                int kk = h & 31;
                *(v8h*)&As[m * KT + kk] =
                    *(const v8h*)&A_src[m * KDIM + k0 + kk];
            }
#endif
            // ---- stage B tile (32xNT fp32 -> quant f16, transposed to [n][k]) ----
            #pragma unroll
            for (int r = 0; r < 2; ++r) {
                int e  = (r * 256 + tid) * 4;          // element index in tile
                int k  = e >> 6;
                int nn = e & 63;
                const float4 w = *(const float4*)&W_src[(size_t)(k0 + k) * NCOLS + n0 + nn];
                float wx = w.x, wy = w.y, wz = w.z, ww = w.w;
                if (phase) {
                    const float4 mk4 = *(const float4*)&Msk_n[(size_t)(k0 + k) * NCOLS + n0 + nn];
                    wx *= mk4.x; wy *= mk4.y; wz *= mk4.z; ww *= mk4.w;
                }
                Bs[(nn + 0) * KT + k] = (_Float16)wq8f(wx);
                Bs[(nn + 1) * KT + k] = (_Float16)wq8f(wy);
                Bs[(nn + 2) * KT + k] = (_Float16)wq8f(wz);
                Bs[(nn + 3) * KT + k] = (_Float16)wq8f(ww);
                if (k0 + KT < KDIM)    // prefetch next weight tile toward L2
                    __builtin_prefetch(&W_src[(size_t)(k0 + KT + k) * NCOLS + n0 + nn], 0, 3);
            }
#if HAVE_TDM
            if (wave == 0) __builtin_amdgcn_s_wait_tensorcnt(0);
#endif
            __syncthreads();

            // ---- A fragments (contiguous K: lane<16 -> K{0..15}, lane>=16 -> K{16..31}) ----
            v16h afrag[2];
            #pragma unroll
            for (int mi = 0; mi < 2; ++mi) {
                int row = wr * 32 + mi * 16 + l16;
                const _Float16* ap = &As[row * KT + lhalf * 16];
                v8h lo = *(const v8h*)ap;          // K base..base+7
                v8h hi = *(const v8h*)(ap + 8);    // K base+8..base+15
                afrag[mi] = __builtin_shufflevector(lo, hi,
                    0,1,2,3,4,5,6,7, 8,9,10,11,12,13,14,15);
            }
            // ---- B fragments (lane holds one column, 16 consecutive K) ----
            v16h bfrag[2];
            #pragma unroll
            for (int ni = 0; ni < 2; ++ni) {
                int col = wc * 32 + ni * 16 + l16;
                bfrag[ni] = *(const v16h*)&Bs[col * KT + lhalf * 16];
            }
            // ---- 4 WMMAs per K step ----
            #pragma unroll
            for (int mi = 0; mi < 2; ++mi)
                #pragma unroll
                for (int ni = 0; ni < 2; ++ni)
                    acc[mi][ni] = __builtin_amdgcn_wmma_f32_16x16x32_f16(
                        false, afrag[mi], false, bfrag[ni],
                        (short)0, acc[mi][ni], false, false);
            __syncthreads();
        }

        // ---- per-phase epilogue: + quantized bias, pact/qp(a12|a13) ----
        const float* bias_n = (phase ? bhh : bih) + (size_t)n * NCOLS;
        const float  aa  = phase ? a13 : a12;
        const float  iaa = phase ? i13 : i12;
        #pragma unroll
        for (int mi = 0; mi < 2; ++mi) {
            #pragma unroll
            for (int ni = 0; ni < 2; ++ni) {
                int col = n0 + wc * 32 + ni * 16 + l16;
                float bq = wq8f(bias_n[col]);
                #pragma unroll
                for (int e = 0; e < 8; ++e) {
                    float v = acc[mi][ni][e] + bq;
                    float qv = qpact8(v, aa, iaa);
                    if (phase == 0) {
                        q1[mi][ni][e] = qv;
                    } else {
                        float g = qpact8(q1[mi][ni][e] + qv, a14, i14);
                        int row = wr * 32 + mi * 16 + 8 * lhalf + e;  // C layout: M = v + 8*(lane>=16)
                        G_n[(size_t)row * NCOLS + col] = g;
                    }
                }
            }
        }
    }
}

// ---------------------------------------------------------------------------
// Kernel 3: pointwise LSTM gate math -> new_h, new_c.
// ---------------------------------------------------------------------------
__global__ void __launch_bounds__(256)
lstm_pointwise_kernel(const float* __restrict__ gates, const float* __restrict__ cx,
                      const float* __restrict__ a3,  const float* __restrict__ a4,
                      const float* __restrict__ a5,  const float* __restrict__ a6,
                      const float* __restrict__ a7,  const float* __restrict__ a8,
                      const float* __restrict__ a9,  const float* __restrict__ a10,
                      const float* __restrict__ a11,
                      float* __restrict__ out_h, float* __restrict__ out_c) {
    int idx = blockIdx.x * blockDim.x + threadIdx.x;      // over NB*BATCH*HID
    int n   = idx >> 17;
    int rem = idx & ((BATCH * KDIM) - 1);
    int m   = rem >> 10;
    int c   = rem & (KDIM - 1);
    size_t base = (size_t)n * BATCH * NCOLS + (size_t)m * NCOLS + c;

    float gi = gates[base];
    float gj = gates[base + 1024];
    float gf = gates[base + 2048];
    float go = gates[base + 3072];

    float c3 = a3[n], c4 = a4[n], c5 = a5[n], c6 = a6[n], c7 = a7[n];
    float c8 = a8[n], c9 = a9[n], c10 = a10[n], c11 = a11[n];

    float forget_g = qpact8(sigmoid_(gf), c3, 1.0f / c3);
    float input_g  = qpact8(sigmoid_(gi), c4, 1.0f / c4);
    float activ    = qpact8(tanhf(gj),    c5, 1.0f / c5);
    float out_g    = qpact8(sigmoid_(go), c6, 1.0f / c6);

    float gated    = qpact8(cx[idx] * forget_g, c7, 1.0f / c7);
    float act_in   = qpact8(input_g * activ,    c8, 1.0f / c8);
    float new_c    = qpact8(gated + act_in,     c9, 1.0f / c9);
    float act_c    = qpact8(tanhf(new_c),       c10, 1.0f / c10);
    float new_h    = qpact8(act_c * out_g,      c11, 1.0f / c11);

    out_h[idx] = new_h;
    out_c[idx] = new_c;
}

// ---------------------------------------------------------------------------
extern "C" void kernel_launch(void* const* d_in, const int* in_sizes, int n_in,
                              void* d_out, int out_size, void* d_ws, size_t ws_size,
                              hipStream_t stream) {
    const float* input     = (const float*)d_in[0];
    const float* hx        = (const float*)d_in[1];
    const float* cx        = (const float*)d_in[2];
    const float* w_mask    = (const float*)d_in[3];
    const float* weight_ih = (const float*)d_in[4];
    const float* weight_hh = (const float*)d_in[5];
    const float* bias_ih   = (const float*)d_in[6];
    const float* bias_hh   = (const float*)d_in[7];
    const float* a1  = (const float*)d_in[8];
    const float* a3  = (const float*)d_in[9];
    const float* a4  = (const float*)d_in[10];
    const float* a5  = (const float*)d_in[11];
    const float* a6  = (const float*)d_in[12];
    const float* a7  = (const float*)d_in[13];
    const float* a8  = (const float*)d_in[14];
    const float* a9  = (const float*)d_in[15];
    const float* a10 = (const float*)d_in[16];
    const float* a11 = (const float*)d_in[17];
    const float* a12 = (const float*)d_in[18];
    const float* a13 = (const float*)d_in[19];
    const float* a14 = (const float*)d_in[20];

    char* ws = (char*)d_ws;
    _Float16* Aih  = (_Float16*)ws;                          // 1 MB
    _Float16* Ahh  = (_Float16*)(ws + (1u << 20));           // 1 MB
    float*    gts  = (float*)(ws + (2u << 20));              // 8 MB

    float* out_h = (float*)d_out;
    float* out_c = out_h + (size_t)NB * BATCH * KDIM;

    quant_inputs_kernel<<<(NB * BATCH * KDIM) / 256, 256, 0, stream>>>(
        input, hx, a1, a11, Aih, Ahh);

    gemm_gates_kernel<<<dim3(NCOLS / NT, NB), 256, 0, stream>>>(
        Aih, Ahh, weight_ih, weight_hh, w_mask, bias_ih, bias_hh,
        a12, a13, a14, gts);

    lstm_pointwise_kernel<<<(NB * BATCH * KDIM) / 256, 256, 0, stream>>>(
        gts, cx, a3, a4, a5, a6, a7, a8, a9, a10, a11, out_h, out_c);
}
